// TemporalMambaEncoder_9320079033274
// MI455X (gfx1250) — matrix operand order repaired and checked by reading, not compiled, not run on hardware
//
#include <hip/hip_runtime.h>

typedef __bf16 bf16;
typedef __attribute__((ext_vector_type(16))) __bf16 bf16x16;
typedef __attribute__((ext_vector_type(8)))  __bf16 bf16x8;
typedef __attribute__((ext_vector_type(8)))  float  f32x8;

// Problem constants
#define BSZ   4
#define T_    256
#define NN    32
#define DM    128
#define LNUM  2
#define S_    16
#define DC_   4
#define DI_   256
#define DTR_  8
#define KK    4
#define RR    16
#define BN_   (BSZ*NN)        // 128
#define MROWS (BN_*T_)        // 32768

// ---------------------------------------------------------------------------
// WMMA GEMM: C[M,N] = epi( A[M,K](bf16 row-major) * Wpk(N,K packed) )
// One wave per 16x(16*NW) output tile: the A fragment is loaded once per
// K-chunk and reused across NW WMMAs (4x arithmetic intensity vs 16x16).
// act: 0=none, 1=exact gelu, 2=tanh.  accum: C += result (residual fuse).
// ---------------------------------------------------------------------------
template<int NW>
__global__ __launch_bounds__(256)
void wmma_gemm_kernel(const bf16* __restrict__ A, const bf16* __restrict__ Wpk,
                      float* __restrict__ C, const float* __restrict__ bias,
                      int Kdim, int Ndim, int act, int accum)
{
    const int wavesPerBlock = blockDim.x >> 5;
    const int waveId  = blockIdx.x * wavesPerBlock + (threadIdx.x >> 5);
    const int lane    = threadIdx.x & 31;
    const int nGroups = (Ndim >> 4) / NW;
    const int mTile   = waveId / nGroups;
    const int nGroup  = waveId % nGroups;
    const int mRow    = mTile * 16 + (lane & 15);
    const int hh      = lane >> 4;
    const int nk      = Kdim >> 5;

    f32x8 acc[NW];
    #pragma unroll
    for (int j = 0; j < NW; ++j) acc[j] = {};

    const bf16* arow  = A + (size_t)mRow * Kdim;
    const bf16* bbase = Wpk + ((size_t)(nGroup * NW) * nk * 32 + lane) * 16;

    for (int kc = 0; kc < nk; ++kc) {
        // next-chunk prefetch -> global_prefetch_b8
        __builtin_prefetch(arow + (kc + 1) * 32, 0, 1);
        // A fragment: lane holds K = {16h..16h+15} of this chunk
        bf16x8 alo = *(const bf16x8*)(arow + kc * 32 + 16 * hh);
        bf16x8 ahi = *(const bf16x8*)(arow + kc * 32 + 16 * hh + 8);
        bf16x16 a;
        #pragma unroll
        for (int i = 0; i < 8; ++i) { a[i] = alo[i]; a[8 + i] = ahi[i]; }
        // NW B fragments share the same A fragment
        #pragma unroll
        for (int j = 0; j < NW; ++j) {
            bf16x16 b = *(const bf16x16*)(bbase + ((size_t)(j * nk + kc) * 32) * 16);
            acc[j] = __builtin_amdgcn_wmma_f32_16x16x32_bf16(false, a, false, b,
                                                             (short)0, acc[j], false, false);
        }
    }

    #pragma unroll
    for (int j = 0; j < NW; ++j) {
        const int col = (nGroup * NW + j) * 16 + (lane & 15);
        const float bv = bias ? bias[col] : 0.0f;
        #pragma unroll
        for (int v = 0; v < 8; ++v) {
            const int row = mTile * 16 + v + 8 * hh;
            float r = acc[j][v] + bv;
            if (act == 1)      r = 0.5f * r * (1.0f + erff(r * 0.70710678118f)); // exact gelu
            else if (act == 2) r = tanhf(r);
            const size_t idx = (size_t)row * Ndim + col;
            if (accum) C[idx] += r; else C[idx] = r;
        }
    }
}

// Pack W(Nsrc x Kdim, f32 row-major) -> bf16 B-fragment layout, zero-pad to Npad rows.
__global__ void pack_weight_kernel(const float* __restrict__ W, bf16* __restrict__ Wpk,
                                   int Nsrc, int Kdim, int Npad)
{
    const int nk = Kdim >> 5;
    const int total = (Npad >> 4) * nk * 32;
    const int gid = blockIdx.x * blockDim.x + threadIdx.x;
    if (gid >= total) return;
    const int lane  = gid & 31;
    const int kc    = (gid >> 5) % nk;
    const int nTile = (gid >> 5) / nk;
    const int n     = nTile * 16 + (lane & 15);
    const int kbase = kc * 32 + 16 * (lane >> 4);
    bf16x16 out;
    #pragma unroll
    for (int i = 0; i < 16; ++i) {
        float v = (n < Nsrc) ? W[(size_t)n * Kdim + kbase + i] : 0.0f;
        out[i] = (bf16)v;
    }
    *(bf16x16*)(Wpk + (size_t)gid * 16) = out;
}

// Pack U_eff: weight row n=(k*16+r), col d -> U[k][d][r] * A_scale[k]; N=64, K=128.
__global__ void pack_u_kernel(const float* __restrict__ U, const float* __restrict__ Asc,
                              bf16* __restrict__ Wpk)
{
    const int nk = DM >> 5;                 // 4
    const int total = (64 >> 4) * nk * 32;  // 512
    const int gid = blockIdx.x * blockDim.x + threadIdx.x;
    if (gid >= total) return;
    const int lane  = gid & 31;
    const int kc    = (gid >> 5) % nk;
    const int nTile = (gid >> 5) / nk;
    const int n     = nTile * 16 + (lane & 15);
    const int k     = n >> 4, r = n & 15;
    const float sc  = Asc[k];
    const int kbase = kc * 32 + 16 * (lane >> 4);
    bf16x16 out;
    #pragma unroll
    for (int i = 0; i < 16; ++i)
        out[i] = (bf16)(U[((size_t)k * DM + (kbase + i)) * RR + r] * sc);
    *(bf16x16*)(Wpk + (size_t)gid * 16) = out;
}

__global__ void cvt_bf16_kernel(const float* __restrict__ src, bf16* __restrict__ dst, size_t n)
{
    size_t gid = (size_t)blockIdx.x * blockDim.x + threadIdx.x;
    if (gid < n) dst[gid] = (bf16)src[gid];
}

// h[m,d] = x[b,t,nn,d], m=(b*NN+nn)*T + t
__global__ void init_h_kernel(const float* __restrict__ x, float* __restrict__ h)
{
    size_t gid = (size_t)blockIdx.x * blockDim.x + threadIdx.x;
    if (gid >= (size_t)MROWS * DM) return;
    const int d = gid & 127;
    const size_t m = gid >> 7;
    const int t = m % T_, bn = m / T_;
    const int b = bn >> 5, nn = bn & 31;
    h[gid] = x[(((size_t)(b * T_ + t)) * NN + nn) * DM + d];
}

// causal depthwise conv (DC=4) over xz[:, :DI] + bias + silu -> xh
__global__ void conv_silu_kernel(const float* __restrict__ xz, const float* __restrict__ cw,
                                 const float* __restrict__ cb, float* __restrict__ xh)
{
    size_t gid = (size_t)blockIdx.x * blockDim.x + threadIdx.x;
    if (gid >= (size_t)MROWS * DI_) return;
    const int c = gid & 255;
    const size_t m = gid >> 8;
    const int t = m % T_;
    const size_t base = (m - t) * 512 + c;   // start of this sequence in xz
    float acc = cb[c];
    #pragma unroll
    for (int j = 0; j < DC_; ++j) {
        const int tt = t - (DC_ - 1) + j;
        if (tt >= 0) acc += xz[base + (size_t)tt * 512] * cw[c * DC_ + j];
    }
    xh[gid] = acc / (1.0f + __expf(-acc));   // silu
}

// dt[m,c] = softplus( xdbl[m,:8] @ dtw[c,:] + dtb[c] )
__global__ void dt_kernel(const float* __restrict__ xdbl, const float* __restrict__ dtw,
                          const float* __restrict__ dtb, float* __restrict__ dt)
{
    size_t gid = (size_t)blockIdx.x * blockDim.x + threadIdx.x;
    if (gid >= (size_t)MROWS * DI_) return;
    const int c = gid & 255;
    const size_t m = gid >> 8;
    const float* xr = xdbl + m * 48;
    float acc = dtb[c];
    #pragma unroll
    for (int r = 0; r < DTR_; ++r) acc += xr[r] * dtw[c * DTR_ + r];
    dt[gid] = (acc > 20.f) ? acc : logf(1.f + __expf(acc));
}

// Selective scan: one block per sequence bn (256 threads = all d), B/C staged in LDS.
// Writes y + Dp*xh back into xh in place.
__global__ __launch_bounds__(256)
void scan_kernel(const float* __restrict__ xdbl, const float* __restrict__ dt,
                 float* __restrict__ xh, const float* __restrict__ A_log,
                 const float* __restrict__ Dp)
{
    const int bn = blockIdx.x;
    const int d  = threadIdx.x;
    __shared__ float sB[S_], sC[S_];
    float Aneg[S_], carry[S_];
    #pragma unroll
    for (int s = 0; s < S_; ++s) { Aneg[s] = -__expf(A_log[d * S_ + s]); carry[s] = 0.f; }
    const float Dv = Dp[d];
    for (int t = 0; t < T_; ++t) {
        const size_t m = (size_t)bn * T_ + t;
        if (threadIdx.x < 32) {
            const float v = xdbl[m * 48 + 8 + threadIdx.x];
            if (threadIdx.x < 16) sB[threadIdx.x] = v; else sC[threadIdx.x - 16] = v;
        }
        __syncthreads();
        const float dtv = dt[m * DI_ + d];
        const float xv  = xh[m * DI_ + d];
        float y = 0.f;
        #pragma unroll
        for (int s = 0; s < S_; ++s) {
            const float dA = __expf(dtv * Aneg[s]);
            carry[s] = dA * carry[s] + dtv * sB[s] * xv;
            y += carry[s] * sC[s];
        }
        __syncthreads();
        xh[m * DI_ + d] = y + Dv * xv;
    }
}

// ygbf[m,c] = bf16( y[m,c] * silu(z[m,c]) ),  z = xz[:, 256+c]
__global__ void gate_y_kernel(const float* __restrict__ y, const float* __restrict__ xz,
                              bf16* __restrict__ out)
{
    size_t gid = (size_t)blockIdx.x * blockDim.x + threadIdx.x;
    if (gid >= (size_t)MROWS * DI_) return;
    const int c = gid & 255;
    const size_t m = gid >> 8;
    const float z = xz[m * 512 + 256 + c];
    out[gid] = (bf16)(y[gid] * (z / (1.f + __expf(-z))));
}

// gate2: logits(5) from gact row, softmax(4) + sigmoid*MAX_DT -> wdt[m][0..4]
__global__ void gate2_kernel(const float* __restrict__ gact, const float* __restrict__ w2,
                             const float* __restrict__ b2, float* __restrict__ wdt)
{
    const int m = blockIdx.x * blockDim.x + threadIdx.x;
    if (m >= MROWS) return;
    const float* r = gact + (size_t)m * DM;
    float lg[5];
    for (int j = 0; j < 5; ++j) {
        float a = b2[j];
        const float* wr = w2 + j * DM;
        for (int d2 = 0; d2 < DM; ++d2) a += r[d2] * wr[d2];
        lg[j] = a;
    }
    const float mx = fmaxf(fmaxf(lg[0], lg[1]), fmaxf(lg[2], lg[3]));
    float e[4], sum = 0.f;
    for (int j = 0; j < 4; ++j) { e[j] = __expf(lg[j] - mx); sum += e[j]; }
    float* o = wdt + (size_t)m * 8;
    for (int j = 0; j < 4; ++j) o[j] = e[j] / sum;
    o[4] = 0.2f / (1.f + __expf(-lg[4]));     // sigmoid * MAX_DT
}

__global__ void mix2_kernel(const float* __restrict__ mact, const float* __restrict__ w2,
                            const float* __restrict__ b2, float* __restrict__ g)
{
    const int m = blockIdx.x * blockDim.x + threadIdx.x;
    if (m >= MROWS) return;
    const float* r = mact + (size_t)m * DM;
    float a = b2[0];
    for (int d2 = 0; d2 < DM; ++d2) a += r[d2] * w2[d2];
    g[m] = 1.f / (1.f + __expf(-a));
}

// hA[m,d] = sum_k w[m,k] * sum_r hU[m,k*16+r] * V[k,r,d]
__global__ void amix_combine_kernel(const float* __restrict__ hU, const float* __restrict__ V,
                                    const float* __restrict__ wdt, float* __restrict__ hA)
{
    size_t gid = (size_t)blockIdx.x * blockDim.x + threadIdx.x;
    if (gid >= (size_t)MROWS * DM) return;
    const int d = gid & 127;
    const size_t m = gid >> 7;
    const float* hu = hU + m * 64;
    const float* wv = wdt + m * 8;
    float acc = 0.f;
    #pragma unroll
    for (int k = 0; k < KK; ++k) {
        float sub = 0.f;
        #pragma unroll
        for (int r = 0; r < RR; ++r)
            sub += hu[k * RR + r] * V[(size_t)((k * RR + r) << 7) + d];
        acc += wv[k] * sub;
    }
    hA[gid] = acc;
}

__global__ void final_kernel(const float* __restrict__ h, const float* __restrict__ hA,
                             const float* __restrict__ hA2, const float* __restrict__ wdt,
                             const float* __restrict__ g, const float* __restrict__ gamma,
                             const float* __restrict__ beta, float* __restrict__ out)
{
    size_t gid = (size_t)blockIdx.x * blockDim.x + threadIdx.x;
    if (gid >= (size_t)MROWS * DM) return;
    const int d = gid & 127;
    const size_t m = gid >> 7;
    const float dtv = wdt[m * 8 + 4];
    float u = tanhf(dtv * hA[gid] + 0.5f * dtv * dtv * hA2[gid]);
    u = gamma[gid] * u + beta[gid];
    const float hd = h[gid] + g[m] * u;
    const int t = m % T_, bn = m / T_;
    const int b = bn >> 5, nn = bn & 31;
    out[(((size_t)(b * T_ + t)) * NN + nn) * DM + d] = hd;
}

// ---------------------------------------------------------------------------
static inline int elgrid(size_t n) { return (int)((n + 255) / 256); }

extern "C" void kernel_launch(void* const* d_in, const int* in_sizes, int n_in,
                              void* d_out, int out_size, void* d_ws, size_t ws_size,
                              hipStream_t stream)
{
    const float* x       = (const float*)d_in[0];
    const float* in_w    = (const float*)d_in[1];
    const float* conv_w  = (const float*)d_in[2];
    const float* conv_b  = (const float*)d_in[3];
    const float* xpj_w   = (const float*)d_in[4];
    const float* dt_w    = (const float*)d_in[5];
    const float* dt_b    = (const float*)d_in[6];
    const float* A_log   = (const float*)d_in[7];
    const float* D_skip  = (const float*)d_in[8];
    const float* out_w   = (const float*)d_in[9];
    const float* U       = (const float*)d_in[10];
    const float* V       = (const float*)d_in[11];
    const float* A_scale = (const float*)d_in[12];
    const float* gw1     = (const float*)d_in[13];
    const float* gb1     = (const float*)d_in[14];
    const float* gw2     = (const float*)d_in[15];
    const float* gb2     = (const float*)d_in[16];
    const float* fgw     = (const float*)d_in[17];
    const float* fgb     = (const float*)d_in[18];
    const float* fbw     = (const float*)d_in[19];
    const float* fbb     = (const float*)d_in[20];
    const float* mw1     = (const float*)d_in[21];
    const float* mb1     = (const float*)d_in[22];
    const float* mw2     = (const float*)d_in[23];
    const float* mb2     = (const float*)d_in[24];
    float* outp = (float*)d_out;

    char* ws = (char*)d_ws;
    size_t off = 0;
    auto take = [&](size_t bytes) -> char* {
        char* p = ws + off;
        off += (bytes + 255) & ~(size_t)255;
        return p;
    };
    float* h    = (float*)take((size_t)MROWS * 128 * 4);   // 16 MB, persistent
    bf16*  hbf  = (bf16*) take((size_t)MROWS * 256 * 2);   // 16 MB, bf16 A operand
    float* xz   = (float*)take((size_t)MROWS * 512 * 4);   // 64 MB (tail reuse below)
    float* xh   = (float*)take((size_t)MROWS * 256 * 4);   // 32 MB (scan writes y in place)
    float* xdbl = (float*)take((size_t)MROWS * 48 * 4);    //  6 MB
    float* dtb  = (float*)take((size_t)MROWS * 256 * 4);   // 32 MB (tail: gamma|beta)
    bf16* inpk  = (bf16*)take(512 * 128 * 2);
    bf16* xpjpk = (bf16*)take(48 * 256 * 2);
    bf16* outpk = (bf16*)take(128 * 256 * 2);
    bf16* g1pk  = (bf16*)take(128 * 128 * 2);
    bf16* fgpk  = (bf16*)take(128 * 128 * 2);
    bf16* fbpk  = (bf16*)take(128 * 128 * 2);
    bf16* m1pk  = (bf16*)take(128 * 128 * 2);
    bf16* upk   = (bf16*)take(64 * 128 * 2);

    // Tail aliases into xz (columns of the 512-wide region)
    float* gact  = xz;                          // M x 128
    float* hU    = xz + (size_t)MROWS * 128;    // M x 64
    float* hA    = xz + (size_t)MROWS * 192;    // M x 128
    float* hA2   = xz + (size_t)MROWS * 320;    // M x 128
    float* wdt   = xz + (size_t)MROWS * 448;    // M x 8
    float* gbuf  = xz + (size_t)MROWS * 456;    // M
    float* gamma = dtb;                         // M x 128
    float* beta  = dtb + (size_t)MROWS * 128;   // M x 128

    const int MT = MROWS / 16;   // 2048 m-tiles
    auto gemm = [&](const bf16* A, const bf16* W, float* C, const float* bias,
                    int K, int N, int act, int accum) {
        if ((N & 63) == 0) {
            const int waves = MT * (N / 64);
            wmma_gemm_kernel<4><<<waves / 8, 256, 0, stream>>>(A, W, C, bias, K, N, act, accum);
        } else { // N == 48
            const int waves = MT * (N / 48);
            wmma_gemm_kernel<3><<<waves / 8, 256, 0, stream>>>(A, W, C, bias, K, N, act, accum);
        }
    };

    init_h_kernel<<<elgrid((size_t)MROWS * 128), 256, 0, stream>>>(x, h);

    // Pack layer-invariant weights
    pack_weight_kernel<<<elgrid(8 * 4 * 32), 256, 0, stream>>>(gw1, g1pk, 128, 128, 128);
    pack_weight_kernel<<<elgrid(8 * 4 * 32), 256, 0, stream>>>(fgw, fgpk, 128, 128, 128);
    pack_weight_kernel<<<elgrid(8 * 4 * 32), 256, 0, stream>>>(fbw, fbpk, 128, 128, 128);
    pack_weight_kernel<<<elgrid(8 * 4 * 32), 256, 0, stream>>>(mw1, m1pk, 128, 128, 128);
    pack_u_kernel<<<elgrid(4 * 4 * 32), 256, 0, stream>>>(U, A_scale, upk);

    for (int l = 0; l < LNUM; ++l) {
        pack_weight_kernel<<<elgrid(32 * 4 * 32), 256, 0, stream>>>(
            in_w + (size_t)l * 512 * 128, inpk, 512, 128, 512);
        pack_weight_kernel<<<elgrid(3 * 8 * 32), 256, 0, stream>>>(
            xpj_w + (size_t)l * 40 * 256, xpjpk, 40, 256, 48);
        pack_weight_kernel<<<elgrid(8 * 8 * 32), 256, 0, stream>>>(
            out_w + (size_t)l * 128 * 256, outpk, 128, 256, 128);

        cvt_bf16_kernel<<<elgrid((size_t)MROWS * 128), 256, 0, stream>>>(h, hbf, (size_t)MROWS * 128);
        gemm(hbf, inpk, xz, nullptr, 128, 512, 0, 0);                       // in_proj
        conv_silu_kernel<<<elgrid((size_t)MROWS * 256), 256, 0, stream>>>(
            xz, conv_w + (size_t)l * 256 * 4, conv_b + (size_t)l * 256, xh);
        cvt_bf16_kernel<<<elgrid((size_t)MROWS * 256), 256, 0, stream>>>(xh, hbf, (size_t)MROWS * 256);
        gemm(hbf, xpjpk, xdbl, nullptr, 256, 48, 0, 0);                     // x_proj
        dt_kernel<<<elgrid((size_t)MROWS * 256), 256, 0, stream>>>(
            xdbl, dt_w + (size_t)l * 256 * 8, dt_b + (size_t)l * 256, dtb);
        scan_kernel<<<BN_, 256, 0, stream>>>(
            xdbl, dtb, xh, A_log + (size_t)l * 256 * 16, D_skip + (size_t)l * 256);
        gate_y_kernel<<<elgrid((size_t)MROWS * 256), 256, 0, stream>>>(xh, xz, hbf);
        gemm(hbf, outpk, h, nullptr, 256, 128, 0, 1);                       // out_proj, h += ...
    }

    // Tail
    cvt_bf16_kernel<<<elgrid((size_t)MROWS * 128), 256, 0, stream>>>(h, hbf, (size_t)MROWS * 128);
    gemm(hbf, g1pk, gact, gb1, 128, 128, 1, 0);                             // gelu(h@gw1+b1)
    gate2_kernel<<<elgrid(MROWS), 256, 0, stream>>>(gact, gw2, gb2, wdt);
    gemm(hbf, m1pk, gact, mb1, 128, 128, 1, 0);                             // gelu(h@mw1+b1)
    mix2_kernel<<<elgrid(MROWS), 256, 0, stream>>>(gact, mw2, mb2, gbuf);
    gemm(hbf, fgpk, gamma, fgb, 128, 128, 2, 0);                            // tanh film gamma
    gemm(hbf, fbpk, beta,  fbb, 128, 128, 2, 0);                            // tanh film beta
    gemm(hbf, upk, hU, nullptr, 128, 64, 0, 0);                             // h @ U_eff
    amix_combine_kernel<<<elgrid((size_t)MROWS * 128), 256, 0, stream>>>(hU, V, wdt, hA);
    cvt_bf16_kernel<<<elgrid((size_t)MROWS * 128), 256, 0, stream>>>(hA, hbf, (size_t)MROWS * 128);
    gemm(hbf, upk, hU, nullptr, 128, 64, 0, 0);                             // hA @ U_eff
    amix_combine_kernel<<<elgrid((size_t)MROWS * 128), 256, 0, stream>>>(hU, V, wdt, hA2);
    final_kernel<<<elgrid((size_t)MROWS * 128), 256, 0, stream>>>(
        h, hA, hA2, wdt, gbuf, gamma, beta, outp);
}
